// GCN_3_layers_31860067401788
// MI455X (gfx1250) — matrix operand order repaired and checked by reading, not compiled, not run on hardware
//
#include <hip/hip_runtime.h>
#include <math.h>

// ---------------------------------------------------------------------------
// 3-layer GCN for MI455X (gfx1250, wave32).
// GEMMs via v_wmma_f32_16x16x32_f16 (f16 in, f32 accumulate), weights staged
// in LDS. Scatter-add over edges via f32 global atomics (bandwidth-bound part).
// ---------------------------------------------------------------------------

typedef _Float16 v16h __attribute__((ext_vector_type(16)));
typedef float    v8f  __attribute__((ext_vector_type(8)));

#define NEG_SLOPE 0.1f

// ---------------- zero fill ----------------
__global__ void gcn_zero_kernel(float* __restrict__ p, long long n) {
  long long i = (long long)blockIdx.x * blockDim.x + threadIdx.x;
  if (i < n) p[i] = 0.0f;
}

// ---------------- degrees ----------------
__global__ void gcn_degree_kernel(const int* __restrict__ src, const int* __restrict__ dst,
                                  float* __restrict__ degO, float* __restrict__ degI, int nE) {
  int e = blockIdx.x * blockDim.x + threadIdx.x;
  if (e < nE) {
    atomicAdd(&degO[src[e]], 1.0f);
    atomicAdd(&degI[dst[e]], 1.0f);
  }
}

__global__ void gcn_rsqrt_kernel(float* __restrict__ d, int n) {
  int i = blockIdx.x * blockDim.x + threadIdx.x;
  if (i < n) {
    float v = d[i];
    v = v < 1.0f ? 1.0f : v;
    d[i] = rsqrtf(v);
  }
}

// ---------------- WMMA GEMM + deg_out row scale ----------------
// H[row, :] = (X[row, :] @ W) * rs_out[row]
// One 16-row tile per wave; all NOUT/16 column tiles per wave.
// ntiles = nrows/16 (nrows is a multiple of 16 -> EXEC all-ones for WMMA).
template<int K, int NOUT>
__global__ __launch_bounds__(256)
void gcn_gemm_rowscale(const float* __restrict__ X, const float* __restrict__ W,
                       const float* __restrict__ rs_out, float* __restrict__ H,
                       int ntiles) {
  // LDS: transposed f16 weights Wt[n][k], contiguous in k per column n.
  __shared__ __align__(32) _Float16 Wt[K * NOUT];
  for (int i = threadIdx.x; i < K * NOUT; i += 256) {
    int n = i / K, k = i - n * K;
    Wt[i] = (_Float16)W[k * NOUT + n];
  }
  __syncthreads();

  const int wave = threadIdx.x >> 5;
  const int lane = threadIdx.x & 31;
  const int rowtile = blockIdx.x * 8 + wave;
  if (rowtile >= ntiles) return;           // uniform per-wave branch
  const int hi = lane >> 4;                // lane half (0/1)
  const int lo = lane & 15;
  const int row = rowtile * 16 + lo;

  constexpr int KS = K / 32;               // k-steps of 32
  constexpr int CT = NOUT / 16;            // 16-col output tiles

  // A fragments (16x32 f16): lanes 0-15 carry K {0..15}, lanes 16-31
  // carry K {16..31} -> two contiguous 8-float chunks per lane.
  v16h afrag[KS];
  const float* xr = X + (size_t)row * K + hi * 16;
  #pragma unroll
  for (int ks = 0; ks < KS; ++ks) {
    const float4* p0 = (const float4*)(xr + ks * 32);
    const float4* p1 = (const float4*)(xr + ks * 32 + 8);
    float4 f0 = p0[0], f1 = p0[1], f2 = p1[0], f3 = p1[1];
    v16h a;
    a[0]  = (_Float16)f0.x; a[1]  = (_Float16)f0.y; a[2]  = (_Float16)f0.z; a[3]  = (_Float16)f0.w;
    a[4]  = (_Float16)f1.x; a[5]  = (_Float16)f1.y; a[6]  = (_Float16)f1.z; a[7]  = (_Float16)f1.w;
    a[8]  = (_Float16)f2.x; a[9]  = (_Float16)f2.y; a[10] = (_Float16)f2.z; a[11] = (_Float16)f2.w;
    a[12] = (_Float16)f3.x; a[13] = (_Float16)f3.y; a[14] = (_Float16)f3.z; a[15] = (_Float16)f3.w;
    afrag[ks] = a;
  }

  v8f acc[CT];
  #pragma unroll
  for (int ct = 0; ct < CT; ++ct) {
    v8f c = {};
    #pragma unroll
    for (int ks = 0; ks < KS; ++ks) {
      // B fragment (32x16 f16): lane half holds 16 contiguous K values for
      // column n = ct*16+lo -> single aligned 32-byte LDS read.
      const v16h b = *(const v16h*)&Wt[(ct * 16 + lo) * K + hi * 16 + ks * 32];
      c = __builtin_amdgcn_wmma_f32_16x16x32_f16(
            /*neg_a=*/false, afrag[ks], /*neg_b=*/false, b,
            /*c_mod=*/(short)0, c, /*reuse_a=*/false, /*reuse_b=*/false);
    }
    acc[ct] = c;
  }

  // D layout: VGPR j -> row m = j + hi*8, col n = ct*16 + lo. Fuse deg_out scale.
  #pragma unroll
  for (int j = 0; j < 8; ++j) {
    int m = rowtile * 16 + j + hi * 8;
    float s = rs_out[m];
    #pragma unroll
    for (int ct = 0; ct < CT; ++ct)
      H[(size_t)m * NOUT + ct * 16 + lo] = acc[ct][j] * s;
  }
}

// ---------------- edge scatter-add ----------------
// agg[dst[e], :] += h[src[e], :]   (F/4 threads per edge, float4 gather)
template<int F>
__global__ __launch_bounds__(256)
void gcn_scatter(const float* __restrict__ H, float* __restrict__ AGG,
                 const int* __restrict__ src, const int* __restrict__ dst, int nE) {
  constexpr int G = F / 4;
  long long idx = (long long)blockIdx.x * blockDim.x + threadIdx.x;
  int e = (int)(idx / G);
  int g = (int)(idx - (long long)e * G);
  if (e >= nE) return;
  int s = src[e], d = dst[e];
  const float4 v = ((const float4*)(H + (size_t)s * F))[g];
  float* a = AGG + (size_t)d * F + g * 4;
  atomicAdd(a + 0, v.x);
  atomicAdd(a + 1, v.y);
  atomicAdd(a + 2, v.z);
  atomicAdd(a + 3, v.w);
}

// ---------------- agg * rsqrt(deg_in) + b, leaky ReLU (in place) -----------
template<int F>
__global__ void gcn_finalize_lrelu(float* __restrict__ AGG, const float* __restrict__ rs_in,
                                   const float* __restrict__ b, int nNodes) {
  long long i = (long long)blockIdx.x * blockDim.x + threadIdx.x;
  if (i >= (long long)nNodes * F) return;
  int row = (int)(i / F);
  int f   = (int)(i - (long long)row * F);
  float v = AGG[i] * rs_in[row] + b[f];
  AGG[i] = v >= 0.0f ? v : NEG_SLOPE * v;
}

// ---------------- final layer: bias + scale + softmax over 32 cols ---------
// One wave32 per row: lane == column. Cross-lane max/sum via shfl_xor.
__global__ __launch_bounds__(256)
void gcn_softmax32(const float* __restrict__ AGG, const float* __restrict__ rs_in,
                   const float* __restrict__ b, float* __restrict__ out, int nNodes) {
  int wave = threadIdx.x >> 5;
  int lane = threadIdx.x & 31;
  int row = blockIdx.x * 8 + wave;
  if (row >= nNodes) return;
  float v = AGG[(size_t)row * 32 + lane] * rs_in[row] + b[lane];
  float m = v;
  #pragma unroll
  for (int o = 16; o >= 1; o >>= 1) m = fmaxf(m, __shfl_xor(m, o, 32));
  float e = __expf(v - m);
  float s = e;
  #pragma unroll
  for (int o = 16; o >= 1; o >>= 1) s += __shfl_xor(s, o, 32);
  out[(size_t)row * 32 + lane] = e / s;
}

// ---------------------------------------------------------------------------

static inline int cdiv_ll(long long a, long long b) { return (int)((a + b - 1) / b); }

extern "C" void kernel_launch(void* const* d_in, const int* in_sizes, int n_in,
                              void* d_out, int out_size, void* d_ws, size_t ws_size,
                              hipStream_t stream) {
  const float* x   = (const float*)d_in[0];
  const int*   src = (const int*)  d_in[1];
  const int*   dst = (const int*)  d_in[2];
  const float* W1  = (const float*)d_in[3];
  const float* b1  = (const float*)d_in[4];
  const float* W2  = (const float*)d_in[5];
  const float* b2  = (const float*)d_in[6];
  const float* W3  = (const float*)d_in[7];
  const float* b3  = (const float*)d_in[8];
  float* out = (float*)d_out;

  const int N  = in_sizes[0] / 128;   // 50000 nodes (multiple of 16)
  const int nE = in_sizes[1];         // 800000 edges
  const int ntiles = N / 16;          // 3125 row tiles

  // Workspace layout (floats): degO[N] | degI[N] | buf0[N*128] | buf1[N*128]
  float* ws   = (float*)d_ws;
  float* degO = ws;
  float* degI = ws + N;
  float* buf0 = degI + N;
  float* buf1 = buf0 + (size_t)N * 128;

  const int T = 256;

  // degrees -> rsqrt(max(deg,1))
  gcn_zero_kernel<<<cdiv_ll(2LL * N, T), T, 0, stream>>>(degO, 2LL * N);
  gcn_degree_kernel<<<cdiv_ll(nE, T), T, 0, stream>>>(src, dst, degO, degI, nE);
  gcn_rsqrt_kernel<<<cdiv_ll(2LL * N, T), T, 0, stream>>>(degO, 2 * N);

  // ---- layer 1: 128 -> 128 ----
  gcn_gemm_rowscale<128, 128><<<cdiv_ll(ntiles, 8), T, 0, stream>>>(x, W1, degO, buf0, ntiles);
  gcn_zero_kernel<<<cdiv_ll((long long)N * 128, T), T, 0, stream>>>(buf1, (long long)N * 128);
  gcn_scatter<128><<<cdiv_ll((long long)nE * 32, T), T, 0, stream>>>(buf0, buf1, src, dst, nE);
  gcn_finalize_lrelu<128><<<cdiv_ll((long long)N * 128, T), T, 0, stream>>>(buf1, degI, b1, N);

  // ---- layer 2: 128 -> 64 ----
  gcn_gemm_rowscale<128, 64><<<cdiv_ll(ntiles, 8), T, 0, stream>>>(buf1, W2, degO, buf0, ntiles);
  gcn_zero_kernel<<<cdiv_ll((long long)N * 64, T), T, 0, stream>>>(buf1, (long long)N * 64);
  gcn_scatter<64><<<cdiv_ll((long long)nE * 16, T), T, 0, stream>>>(buf0, buf1, src, dst, nE);
  gcn_finalize_lrelu<64><<<cdiv_ll((long long)N * 64, T), T, 0, stream>>>(buf1, degI, b2, N);

  // ---- layer 3: 64 -> 32, then softmax ----
  gcn_gemm_rowscale<64, 32><<<cdiv_ll(ntiles, 8), T, 0, stream>>>(buf1, W3, degO, buf0, ntiles);
  gcn_zero_kernel<<<cdiv_ll((long long)N * 32, T), T, 0, stream>>>(buf1, (long long)N * 32);
  gcn_scatter<32><<<cdiv_ll((long long)nE * 8, T), T, 0, stream>>>(buf0, buf1, src, dst, nE);
  gcn_softmax32<<<cdiv_ll(N, 8), T, 0, stream>>>(buf1, degI, b3, out, N);
}
